// NetVLAD_13262859010458
// MI455X (gfx1250) — matrix operand order repaired and checked by reading, not compiled, not run on hardware
//
#include <hip/hip_runtime.h>

// ---------------------------------------------------------------------------
// NetVLAD fused pipeline for gfx1250 (MI455X), wave32 WMMA bf16 path.
// B=32, H*W=N=1024, D=512, K=64.
// ---------------------------------------------------------------------------

typedef __attribute__((ext_vector_type(16))) __bf16 v16bf;
typedef __attribute__((ext_vector_type(8)))  __bf16 v8bf;
typedef __attribute__((ext_vector_type(4)))  __bf16 v4bf;
typedef __attribute__((ext_vector_type(8)))  float  v8f;

#define NV_EPS 1e-12f

__device__ __forceinline__ __bf16 f2bf(float f) {
  // round-to-nearest-even f32 -> bf16
  unsigned u = __float_as_uint(f);
  u += 0x7FFFu + ((u >> 16) & 1u);
  unsigned short h = (unsigned short)(u >> 16);
  return __builtin_bit_cast(__bf16, h);
}

// ---------------------------------------------------------------------------
// Kernel 0: repack Wk (512x64 f32, row-major [d][k]) into WMMA B-fragment
// order as bf16. Fragment fi = kt*4+nt covers d in [32kt,32kt+32),
// k in [16nt,16nt+16). Contiguous-k 16-bit B 32x16 layout (same k order as the A loads):
//   lane l holds column k = 16nt + (l&15); element e holds row
//   d = 32kt + e + 16*(l>=16).
// Each fragment is 32 lanes x 16 bf16 contiguous -> coalesced 32B/lane loads.
// ---------------------------------------------------------------------------
__global__ __launch_bounds__(256) void netvlad_prep_wk(
    const float* __restrict__ Wk, __bf16* __restrict__ wkfrag) {
  int base = blockIdx.x * 256 + threadIdx.x;      // grid 32 -> 8192 threads
  for (int rep = 0; rep < 4; ++rep) {
    int idx = base + rep * 8192;                  // 0 .. 32767
    int fi = idx >> 9;                            // fragment (64 total)
    int r  = idx & 511;
    int l  = r >> 4;                              // lane 0..31
    int e  = r & 15;                              // element 0..15
    int kt = fi >> 2, nt = fi & 3;
    int d  = kt * 32 + e + ((l >= 16) ? 16 : 0);
    int kc = nt * 16 + (l & 15);
    wkfrag[idx] = f2bf(Wk[d * 64 + kc]);
  }
}

// ---------------------------------------------------------------------------
// Kernel 1: fused 1x1-conv + softmax + VLAD aggregation.
// grid = 32 batches x 16 row-splits (64 rows each), block = 512 (16 waves).
//
// LDS plan (91,136 B total, aliased across phases):
//   xa : bf16[64][520]  @0       (66,560 B)   x tile, n-major, padded rows
//   sA : f32 [64][64]   @66,560  (16,384 B)   logits, skew-addressed
//   xt : bf16[512][72]  @0       (73,728 B)   x tile transposed (aliases xa+sA,
//                                              built after softmax consumed s)
//   af : bf16[8][32][16]@82,944  ( 8,192 B)   'a' in B-fragment order
// ---------------------------------------------------------------------------
__global__ __launch_bounds__(512) void netvlad_main(
    const float* __restrict__ x, const __bf16* __restrict__ wkfrag,
    const float* __restrict__ bias, float* __restrict__ asum_g,
    float* __restrict__ vacc_g) {
  __shared__ __align__(64) unsigned char smem[91136];
  __bf16* xa = (__bf16*)(smem);
  float*  sA = (float*)(smem + 66560);
  __bf16* xt = (__bf16*)(smem);
  __bf16* af = (__bf16*)(smem + 82944);

  const int t  = threadIdx.x;
  const int w  = t >> 5;        // wave 0..15
  const int l  = t & 31;        // lane
  const int ll = l & 15;
  const int hi = l >> 4;        // lane half

  const int b  = blockIdx.x >> 4;   // batch
  const int sp = blockIdx.x & 15;   // row split
  const float* xb = x + ((size_t)(b * 1024 + sp * 64)) * 512;

  // ---- Phase 0: stream x tile (64x512 f32) -> xa bf16 [n][d], coalesced ----
  for (int it = 0; it < 16; ++it) {
    int i4 = t + it * 512;
    int n  = i4 >> 7;
    int d  = (i4 & 127) << 2;
    float4 v = *(const float4*)(xb + n * 512 + d);
    v4bf p;
    p[0] = f2bf(v.x); p[1] = f2bf(v.y); p[2] = f2bf(v.z); p[3] = f2bf(v.w);
    *(v4bf*)(xa + n * 520 + d) = p;
  }
  __syncthreads();

  // ---- Phase 1: GEMM1  s(64x64) = X(64x512) @ Wk(512x64), bf16 WMMA -------
  {
    const int mt = w & 3, nt = w >> 2;   // 16 waves -> 16 output tiles
    v8f acc = {0.f, 0.f, 0.f, 0.f, 0.f, 0.f, 0.f, 0.f};
    const int n = mt * 16 + ll;
    for (int kt = 0; kt < 16; ++kt) {
      const int k0 = kt * 32;
      // A 16x32 bf16 fragment: run0 K = k0+16*hi.. , run1 K = k0+16*hi+8..
      v8bf r0 = *(const v8bf*)(xa + n * 520 + k0 + hi * 16);
      v8bf r1 = *(const v8bf*)(xa + n * 520 + k0 + hi * 16 + 8);
      v16bf A;
#pragma unroll
      for (int i = 0; i < 8; ++i) { A[i] = r0[i]; A[i + 8] = r1[i]; }
      v16bf Bf = *(const v16bf*)(wkfrag + (kt * 4 + nt) * 512 + l * 16);
      acc = __builtin_amdgcn_wmma_f32_16x16x32_bf16(false, A, false, Bf,
                                                    (short)0, acc, false, false);
    }
    const int k  = nt * 16 + ll;
    const float bk = bias[k];
#pragma unroll
    for (int j = 0; j < 8; ++j) {
      const int n2 = mt * 16 + j + hi * 8;          // C/D layout: M=j / M=j+8
      sA[n2 * 64 + ((k + n2) & 63)] = acc[j] + bk;  // skewed, bank-conflict-free
    }
  }
  __syncthreads();

  // ---- Phase 2: row softmax over K=64; write a into B-frag order ----------
  if (t < 64) {
    const int r = t;
    float mx = -1e30f;
    for (int j = 0; j < 64; ++j)
      mx = fmaxf(mx, sA[r * 64 + ((j + r) & 63)]);
    float sum = 0.f;
    for (int j = 0; j < 64; ++j) {
      float e = __expf(sA[r * 64 + ((j + r) & 63)] - mx);
      sA[r * 64 + ((j + r) & 63)] = e;
      sum += e;
    }
    const float inv = 1.f / sum;
    const int f0  = (r >= 32) ? 4 : 0;
    const int hib = (((r & 31) >= 16) ? 16 : 0);
    const int e0  = r & 15;
    for (int j = 0; j < 64; ++j) {
      float a = sA[r * 64 + ((j + r) & 63)] * inv;
      sA[r * 64 + ((j + r) & 63)] = a;   // keep f32 copy for column sums
      af[(f0 + (j >> 4)) * 512 + ((j & 15) + hib) * 16 + e0] = f2bf(a);
    }
  }
  __syncthreads();

  // ---- Phase 3: partial column sums of a -> global asum (fp32 atomics) ----
  if (t < 64) {
    float s = 0.f;
    for (int n = 0; n < 64; ++n) s += sA[n * 64 + ((t + n) & 63)];
    unsafeAtomicAdd(asum_g + b * 64 + t, s);
  }
  __syncthreads();

  // ---- Phase 4: rebuild x tile transposed: xt[d][n] (aliases xa/sA) -------
  // x is L2-resident by now; re-read is cheap and avoids 165 KB LDS.
  for (int it = 0; it < 16; ++it) {
    int i4 = t + it * 512;
    int n  = i4 >> 7;
    int d  = (i4 & 127) << 2;
    float4 v = *(const float4*)(xb + n * 512 + d);
    xt[(d + 0) * 72 + n] = f2bf(v.x);
    xt[(d + 1) * 72 + n] = f2bf(v.y);
    xt[(d + 2) * 72 + n] = f2bf(v.z);
    xt[(d + 3) * 72 + n] = f2bf(v.w);
  }
  __syncthreads();

  // ---- Phase 5: GEMM2  v(512x64) += xT(512x64n) @ a(64nx64), bf16 WMMA ----
  {
    const int dt0 = w * 2;                 // each wave: 2 D-tiles x 4 K-tiles
    v8f acc[2][4] = {};
    for (int ns = 0; ns < 2; ++ns) {       // reduction over n in 2 steps of 32
      const int n0 = ns * 32;
      v16bf Bf[4];
#pragma unroll
      for (int kt = 0; kt < 4; ++kt)
        Bf[kt] = *(const v16bf*)(af + (ns * 4 + kt) * 512 + l * 16);
#pragma unroll
      for (int dd = 0; dd < 2; ++dd) {
        const int d = (dt0 + dd) * 16 + ll;
        v8bf r0 = *(const v8bf*)(xt + d * 72 + n0 + hi * 16);
        v8bf r1 = *(const v8bf*)(xt + d * 72 + n0 + hi * 16 + 8);
        v16bf A;
#pragma unroll
        for (int i = 0; i < 8; ++i) { A[i] = r0[i]; A[i + 8] = r1[i]; }
#pragma unroll
        for (int kt = 0; kt < 4; ++kt)
          acc[dd][kt] = __builtin_amdgcn_wmma_f32_16x16x32_bf16(
              false, A, false, Bf[kt], (short)0, acc[dd][kt], false, false);
      }
    }
    // Accumulate partial v into L2-resident global buffer (fp32 atomics).
    float* vb = vacc_g + (size_t)b * 32768;
#pragma unroll
    for (int dd = 0; dd < 2; ++dd)
#pragma unroll
      for (int kt = 0; kt < 4; ++kt)
#pragma unroll
        for (int j = 0; j < 8; ++j) {
          const int d = (dt0 + dd) * 16 + j + hi * 8;
          const int k = kt * 16 + ll;
          unsafeAtomicAdd(vb + d * 64 + k, acc[dd][kt][j]);
        }
  }
}

// ---------------------------------------------------------------------------
// Kernel 2: finalize. v = vacc + asum[k]*C[d][k]; intra-L2 over D per k;
// flat norm computed analytically: F = sum_k ss_k/(ss_k+eps).
// grid = 32 (one block per batch), block = 256: thread owns column k=t&63,
// rows d = (t>>6) + 4j.
// ---------------------------------------------------------------------------
__global__ __launch_bounds__(256) void netvlad_finalize(
    const float* __restrict__ vacc, const float* __restrict__ asum_g,
    const float* __restrict__ C, float* __restrict__ out) {
  __shared__ float ss[64];
  __shared__ float fl[64];
  const int b = blockIdx.x;
  const int t = threadIdx.x;
  const int k = t & 63;
  const int d0 = t >> 6;

  if (t < 64) ss[t] = 0.f;
  __syncthreads();

  const float* vb = vacc + (size_t)b * 32768;
  const float  as = asum_g[b * 64 + k];

  float local = 0.f;
  for (int j = 0; j < 128; ++j) {
    const int d = d0 + j * 4;
    const float v = vb[d * 64 + k] + as * C[d * 64 + k];
    local += v * v;
  }
  atomicAdd(&ss[k], local);           // ds_add_f32
  __syncthreads();

  if (t < 64) fl[t] = ss[t] / (ss[t] + NV_EPS);
  __syncthreads();

  float F = 0.f;
  for (int j = 0; j < 64; ++j) F += fl[j];   // broadcast LDS reads

  const float scale = (1.f / sqrtf(ss[k] + NV_EPS)) * (1.f / sqrtf(F + NV_EPS));
  float* ob = out + (size_t)b * 32768;
  for (int j = 0; j < 128; ++j) {
    const int d = d0 + j * 4;
    const float v = vb[d * 64 + k] + as * C[d * 64 + k];
    ob[d * 64 + k] = v * scale;
  }
}

// ---------------------------------------------------------------------------
// Host-side launcher. Inputs: x(32*32*32*512 f32), Wk(512*64 f32), b(64 f32),
// C(512*64 f32). Output: (32, 32768) f32.
// Workspace: [wkfrag bf16 64KB][asum f32 8KB][vacc f32 4MB]  (~4.07 MB)
// ---------------------------------------------------------------------------
extern "C" void kernel_launch(void* const* d_in, const int* in_sizes, int n_in,
                              void* d_out, int out_size, void* d_ws, size_t ws_size,
                              hipStream_t stream) {
  (void)in_sizes; (void)n_in; (void)out_size; (void)ws_size;
  const float* x    = (const float*)d_in[0];
  const float* Wk   = (const float*)d_in[1];
  const float* bias = (const float*)d_in[2];
  const float* C    = (const float*)d_in[3];
  float* out = (float*)d_out;

  __bf16* wkfrag = (__bf16*)d_ws;
  float*  asum   = (float*)((char*)d_ws + 65536);
  float*  vacc   = (float*)((char*)d_ws + 65536 + 8192);

  // zero the accumulators (workspace is poisoned by the harness)
  hipMemsetAsync((char*)d_ws + 65536, 0, 8192 + (size_t)32 * 512 * 64 * 4, stream);

  netvlad_prep_wk<<<32, 256, 0, stream>>>(Wk, wkfrag);
  netvlad_main<<<512, 512, 0, stream>>>(x, wkfrag, bias, asum, vacc);
  netvlad_finalize<<<32, 256, 0, stream>>>(vacc, asum, C, out);
}
